// GKCN_85083302133776
// MI455X (gfx1250) — matrix operand order repaired and checked by reading, not compiled, not run on hardware
//
#include <hip/hip_runtime.h>
#include <hip/hip_bf16.h>

// ---------------------------------------------------------------------------
// GraKN 3-layer GNN for MI455X (gfx1250, wave32, WMMA).
//
// Each 16-edge/16-node tile -> v_wmma_f32_16x16x32_bf16 (bf16 A/B, f32 acc).
// - Weights pre-packed to B-fragment lane order (bf16): 2x global_load_b128
//   per fragment, no predication.
// - Gathers use lane-pair contiguous half-rows (compile-time strides) so the
//   compiler emits b64/b128 vector memory ops, no integer-division math.
// - Tiles are wave-private: only compiler wave barriers, no s_barrier traffic.
// - Contiguous edge-attr rows staged with CDNA5 async copy
//   (global_load_async_to_lds_b128 / ASYNCcnt).
// ---------------------------------------------------------------------------

#define DEVINL __device__ __forceinline__

typedef __attribute__((ext_vector_type(16))) __bf16 v16bf;
typedef __attribute__((ext_vector_type(8)))  float  v8f;

static constexpr int N_NODES = 100000;
static constexpr int N_EDGES = 3200000;

DEVINL v8f wmma_bf16(v16bf a, v16bf b, v8f c) {
  return __builtin_amdgcn_wmma_f32_16x16x32_bf16(false, a, false, b, (short)0, c,
                                                 false, false);
}

// Intra-wave LDS ordering only: same-wave DS ops are in-order (ISA), we just
// need to stop the compiler reordering across the producer/consumer point.
DEVINL void wave_sync() { __builtin_amdgcn_wave_barrier(); }

// A fragment (16x32 bf16) from an LDS tile, row-major with stride S.
// Lanes 0-15 -> rows M=0..15, K=0..7 in v0..3 pairs, K=16..23 in v4..7;
// lanes 16-31 -> same rows, K=8..15 / K=24..31.
DEVINL v16bf frag_A_lds(const float* T, int S, int lane, int kbase, int klim) {
  const int m  = lane & 15;
  const int kh = (lane >> 4) << 3;  // 0 or 8
  v16bf a;
#pragma unroll
  for (int v = 0; v < 8; ++v) {
    const int kk = (v < 4) ? (kh + 2 * v) : (16 + kh + 2 * (v - 4));
    const int k0 = kbase + kk;
    a[2 * v]     = (k0     < klim) ? (__bf16)T[m * S + k0]     : (__bf16)0.0f;
    a[2 * v + 1] = (k0 + 1 < klim) ? (__bf16)T[m * S + k0 + 1] : (__bf16)0.0f;
  }
  return a;
}

DEVINL void atomic_add_f32(float* p, float v) { unsafeAtomicAdd(p, v); }

// ---------------------------------------------------------------------------
// Pack one 32-K-slice of fp32 W[K x NW] (row-major) into WMMA B-fragment lane
// order (bf16, zero-padded). 32 threads. Lanes 0-15 hold K=kbase+0..7,16..23
// (col n = lane), lanes 16-31 hold K=kbase+8..15,24..31.
// ---------------------------------------------------------------------------
__global__ void pack_B_kernel(const float* __restrict__ W, int K, int NW,
                              int kbase, __bf16* __restrict__ out) {
  const int lane = threadIdx.x & 31;
  const int n  = lane & 15;
  const int kh = (lane >> 4) << 3;
#pragma unroll
  for (int v = 0; v < 8; ++v) {
    const int k = kbase + ((v < 4) ? (kh + 2 * v) : (16 + kh + 2 * (v - 4)));
    const float w0 = (k     < K && n < NW) ? W[(size_t)k * NW + n]       : 0.0f;
    const float w1 = (k + 1 < K && n < NW) ? W[(size_t)(k + 1) * NW + n] : 0.0f;
    out[lane * 16 + 2 * v]     = (__bf16)w0;
    out[lane * 16 + 2 * v + 1] = (__bf16)w1;
  }
}

// ---------------------------------------------------------------------------
// Edge conv: edge_repr = MLP([x_src | e | x_dst]); scatter [x_src|edge_repr]
// onto agg[dst]. One 16-edge tile per wave, 8 waves/block.
// Node feature stride == DX, edge feature stride == DE (compile-time).
// ---------------------------------------------------------------------------
template <int DX, int DE, int KPAD, int NL, bool LAST_ACT, int OUTD>
__global__ __launch_bounds__(256) void edge_conv_kernel(
    const float* __restrict__ xn, const float* __restrict__ xe,
    const int* __restrict__ src, const int* __restrict__ dst,
    const v16bf* __restrict__ P0, const float* __restrict__ B0,
    const v16bf* __restrict__ P1, const float* __restrict__ B1,
    const v16bf* __restrict__ P2, const float* __restrict__ B2,
    float* __restrict__ agg, float* __restrict__ eout, int numTiles) {
  constexpr int KIN  = DX + DE + DX;
  constexpr int S    = KPAD + 16;   // input cols + hidden scratch cols
  constexpr int HOFF = KPAD;
  constexpr int AGS  = 32;
  constexpr int PADW = KPAD - KIN;  // zero strip width (even: 2 or 16)

  __shared__ __align__(16) float tileBuf[8 * 16 * S];
  __shared__ int sid[8][2][16];

  const int  lane   = threadIdx.x & 31;
  const int  wave   = threadIdx.x >> 5;
  const int  tileId = blockIdx.x * 8 + wave;
  const bool active = tileId < numTiles;
  float* T    = &tileBuf[wave * 16 * S];
  int*   Ssrc = sid[wave][0];
  int*   Sdst = sid[wave][1];

  if (active && lane < 16) {
    Ssrc[lane] = src[tileId * 16 + lane];
    Sdst[lane] = dst[tileId * 16 + lane];
  }
  // zero only the KIN..KPAD pad strip (h region is fully overwritten later)
  const int gm = lane >> 1;               // lane pair -> row
  if constexpr (PADW > 0) {
    const int zc = (lane & 1) * (PADW / 2);
#pragma unroll
    for (int k = 0; k < PADW / 2; ++k) T[gm * S + KIN + zc + k] = 0.0f;
  }
  wave_sync();

  if (active) {
    const int gc = (lane & 1) * (DX / 2);
    const int s0 = Ssrc[gm], d0 = Sdst[gm];
#pragma unroll
    for (int k = 0; k < DX / 2; ++k) {  // contiguous per-lane runs -> b64/b128
      T[gm * S + gc + k]           = xn[(size_t)s0 * DX + gc + k];
      T[gm * S + DX + DE + gc + k] = xn[(size_t)d0 * DX + gc + k];
    }
    if constexpr (DE == 16) {
      // Contiguous 16x16 fp32 block: CDNA5 async copy memory -> LDS.
      const char* gbase = (const char*)(xe + (size_t)tileId * 16 * DE);
#pragma unroll
      for (int it = 0; it < 2; ++it) {
        const int i  = it * 32 + lane;  // 0..63, 16B each
        const int m  = i >> 2;
        const int c4 = (i & 3) << 2;
        const unsigned ldsoff = (unsigned)(uintptr_t)&T[m * S + DX + c4];
        const void* g = gbase + ((size_t)m * DE + c4) * 4;
        asm volatile("global_load_async_to_lds_b128 %0, %1, off"
                     :: "v"(ldsoff), "v"(g) : "memory");
      }
    } else {
      const int ec = (lane & 1) * (DE / 2);
#pragma unroll
      for (int k = 0; k < DE / 2; ++k)
        T[gm * S + DX + ec + k] = xe[(size_t)(tileId * 16 + gm) * DE + ec + k];
    }
  }
  if constexpr (DE == 16) {
    asm volatile("s_wait_asynccnt 0x0" ::: "memory");
  }
  wave_sync();

  const int n    = lane & 15;
  const int moff = (lane >> 4) << 3;

  // ---- layer 0 (always ReLU) ----
  v8f acc = {};
#pragma unroll
  for (int kb = 0; kb < KPAD; kb += 32) {
    v16bf a = frag_A_lds(T, S, lane, kb, KPAD);
    v16bf b = P0[(kb >> 5) * 32 + lane];
    acc = wmma_bf16(a, b, acc);
  }
  {
    const float bias = B0[n];
#pragma unroll
    for (int r = 0; r < 8; ++r) {
      float v = acc[r] + bias;
      T[(r + moff) * S + HOFF + n] = v > 0.0f ? v : 0.0f;
    }
  }
  wave_sync();

  // ---- layer 1 ----
  v8f accF = {};
  {
    v16bf a = frag_A_lds(T + HOFF, S, lane, 0, 16);
    v16bf b = P1[lane];
    accF = wmma_bf16(a, b, accF);
  }
  if constexpr (NL == 3) {  // hidden ReLU + layer 2
    const float bias = B1[n];
#pragma unroll
    for (int r = 0; r < 8; ++r) {
      float v = accF[r] + bias;
      T[(r + moff) * S + HOFF + n] = v > 0.0f ? v : 0.0f;
    }
    wave_sync();
    v8f acc2 = {};
    v16bf a = frag_A_lds(T + HOFF, S, lane, 0, 16);
    v16bf b = P2[lane];
    accF = wmma_bf16(a, b, acc2);
  }

  const float* Bf  = (NL == 3) ? B2 : B1;
  const float bias = Bf[n < OUTD ? n : OUTD - 1];  // clamped, unpredicated
  float outv[8];
#pragma unroll
  for (int r = 0; r < 8; ++r) {
    float v = accF[r] + bias;
    if (LAST_ACT) v = v > 0.0f ? v : 0.0f;
    outv[r] = v;
  }

  if (active && n < OUTD) {
#pragma unroll
    for (int r = 0; r < 8; ++r) {
      const int m = r + moff;
      eout[(size_t)(tileId * 16 + m) * OUTD + n] = outv[r];
      atomic_add_f32(&agg[(size_t)Sdst[m] * AGS + DX + n], outv[r]);
    }
  }
  if (active) {  // scatter x_src (still intact in T cols 0..DX)
    const int gc = (lane & 1) * (DX / 2);
    const int d0 = Sdst[gm];
#pragma unroll
    for (int k = 0; k < DX / 2; ++k)
      atomic_add_f32(&agg[(size_t)d0 * AGS + gc + k], T[gm * S + gc + k]);
  }
}

// ---------------------------------------------------------------------------
// Node conv: x' = MLP(agg). One 16-node tile per wave. agg rows are 32 floats
// with cols >= KIN guaranteed zero, so we load full rows (free zero-padding).
// ---------------------------------------------------------------------------
template <int KIN, int NL, bool LAST_ACT, int OUTD>
__global__ __launch_bounds__(256) void node_conv_kernel(
    const float* __restrict__ agg,
    const v16bf* __restrict__ P0, const float* __restrict__ B0,
    const v16bf* __restrict__ P1, const float* __restrict__ B1,
    const v16bf* __restrict__ P2, const float* __restrict__ B2,
    float* __restrict__ out, int OS, int numTiles) {
  constexpr int KPAD = 32;
  constexpr int S    = KPAD + 16;
  constexpr int HOFF = KPAD;
  constexpr int AGS  = 32;

  __shared__ __align__(16) float tileBuf[8 * 16 * S];

  const int  lane   = threadIdx.x & 31;
  const int  wave   = threadIdx.x >> 5;
  const int  tileId = blockIdx.x * 8 + wave;
  const bool active = tileId < numTiles;
  float* T = &tileBuf[wave * 16 * S];

  if (active) {
    const int gm = lane >> 1;
    const int gc = (lane & 1) * 16;
#pragma unroll
    for (int k = 0; k < 16; ++k)  // contiguous -> 4x b128 per lane
      T[gm * S + gc + k] = agg[(size_t)(tileId * 16 + gm) * AGS + gc + k];
  }
  wave_sync();

  const int n    = lane & 15;
  const int moff = (lane >> 4) << 3;

  v8f acc = {};
  {
    v16bf a = frag_A_lds(T, S, lane, 0, KPAD);
    v16bf b = P0[lane];
    acc = wmma_bf16(a, b, acc);
  }
  {
    const float bias = B0[n];
#pragma unroll
    for (int r = 0; r < 8; ++r) {
      float v = acc[r] + bias;
      T[(r + moff) * S + HOFF + n] = v > 0.0f ? v : 0.0f;
    }
  }
  wave_sync();

  v8f accF = {};
  {
    v16bf a = frag_A_lds(T + HOFF, S, lane, 0, 16);
    v16bf b = P1[lane];
    accF = wmma_bf16(a, b, accF);
  }
  if constexpr (NL == 3) {
    const float bias = B1[n];
#pragma unroll
    for (int r = 0; r < 8; ++r) {
      float v = accF[r] + bias;
      T[(r + moff) * S + HOFF + n] = v > 0.0f ? v : 0.0f;
    }
    wave_sync();
    v8f acc2 = {};
    v16bf a = frag_A_lds(T + HOFF, S, lane, 0, 16);
    v16bf b = P2[lane];
    accF = wmma_bf16(a, b, acc2);
  }

  const float* Bf  = (NL == 3) ? B2 : B1;
  const float bias = Bf[n < OUTD ? n : OUTD - 1];
  if (active && n < OUTD) {
#pragma unroll
    for (int r = 0; r < 8; ++r) {
      float v = accF[r] + bias;
      if (LAST_ACT) v = v > 0.0f ? v : 0.0f;
      out[(size_t)(tileId * 16 + r + moff) * OS + n] = v;
    }
  }
}

// ---------------------------------------------------------------------------
extern "C" void kernel_launch(void* const* d_in, const int* in_sizes, int n_in,
                              void* d_out, int out_size, void* d_ws,
                              size_t ws_size, hipStream_t stream) {
  (void)in_sizes; (void)n_in; (void)out_size; (void)ws_size;

  const float* x_node = (const float*)d_in[0];  // N x 12
  const float* x_edge = (const float*)d_in[1];  // E x 6
  const int*   ei     = (const int*)d_in[2];    // 2 x E
  const int*   src    = ei;
  const int*   dst    = ei + N_EDGES;

  int p = 3;
  const float* e0w0 = (const float*)d_in[p++]; const float* e0b0 = (const float*)d_in[p++];
  const float* e0w1 = (const float*)d_in[p++]; const float* e0b1 = (const float*)d_in[p++];
  const float* n0w0 = (const float*)d_in[p++]; const float* n0b0 = (const float*)d_in[p++];
  const float* n0w1 = (const float*)d_in[p++]; const float* n0b1 = (const float*)d_in[p++];
  const float* e1w0 = (const float*)d_in[p++]; const float* e1b0 = (const float*)d_in[p++];
  const float* e1w1 = (const float*)d_in[p++]; const float* e1b1 = (const float*)d_in[p++];
  const float* n1w0 = (const float*)d_in[p++]; const float* n1b0 = (const float*)d_in[p++];
  const float* n1w1 = (const float*)d_in[p++]; const float* n1b1 = (const float*)d_in[p++];
  const float* e2w0 = (const float*)d_in[p++]; const float* e2b0 = (const float*)d_in[p++];
  const float* e2w1 = (const float*)d_in[p++]; const float* e2b1 = (const float*)d_in[p++];
  const float* e2w2 = (const float*)d_in[p++]; const float* e2b2 = (const float*)d_in[p++];
  const float* n2w0 = (const float*)d_in[p++]; const float* n2b0 = (const float*)d_in[p++];
  const float* n2w1 = (const float*)d_in[p++]; const float* n2b1 = (const float*)d_in[p++];
  const float* n2w2 = (const float*)d_in[p++]; const float* n2b2 = (const float*)d_in[p++];

  // ---- workspace layout ----
  __bf16* pk = (__bf16*)d_ws;   // 16 packed chunks x 512 bf16 = 16 KB
  auto chunk = [&](int c) { return pk + (size_t)c * 512; };
  float* fws = (float*)((char*)d_ws + 16 * 1024);
  float* xn0 = fws;                              // N x 16
  float* xn1 = xn0 + (size_t)N_NODES * 16;       // N x 16
  float* agg = xn1 + (size_t)N_NODES * 16;       // N x 32
  float* xe0 = agg + (size_t)N_NODES * 32;       // E x 16
  float* xe1 = xe0 + (size_t)N_EDGES * 16;       // E x 16

  float* out_node = (float*)d_out;                         // N x 3
  float* out_edge = (float*)d_out + (size_t)N_NODES * 3;   // E x 3

  // chunk map:
  //  0: e0w0(K30)   1: e0w1       2: n0w0(K28)   3: n0w1
  //  4,5: e1w0(K48) 6: e1w1       7: n1w0(K32)   8: n1w1
  //  9,10: e2w0     11: e2w1      12: e2w2(NW3)  13: n2w0(K19)
  //  14: n2w1       15: n2w2(NW3)
  pack_B_kernel<<<1, 32, 0, stream>>>(e0w0, 30, 16, 0,  chunk(0));
  pack_B_kernel<<<1, 32, 0, stream>>>(e0w1, 16, 16, 0,  chunk(1));
  pack_B_kernel<<<1, 32, 0, stream>>>(n0w0, 28, 16, 0,  chunk(2));
  pack_B_kernel<<<1, 32, 0, stream>>>(n0w1, 16, 16, 0,  chunk(3));
  pack_B_kernel<<<1, 32, 0, stream>>>(e1w0, 48, 16, 0,  chunk(4));
  pack_B_kernel<<<1, 32, 0, stream>>>(e1w0, 48, 16, 32, chunk(5));
  pack_B_kernel<<<1, 32, 0, stream>>>(e1w1, 16, 16, 0,  chunk(6));
  pack_B_kernel<<<1, 32, 0, stream>>>(n1w0, 32, 16, 0,  chunk(7));
  pack_B_kernel<<<1, 32, 0, stream>>>(n1w1, 16, 16, 0,  chunk(8));
  pack_B_kernel<<<1, 32, 0, stream>>>(e2w0, 48, 16, 0,  chunk(9));
  pack_B_kernel<<<1, 32, 0, stream>>>(e2w0, 48, 16, 32, chunk(10));
  pack_B_kernel<<<1, 32, 0, stream>>>(e2w1, 16, 16, 0,  chunk(11));
  pack_B_kernel<<<1, 32, 0, stream>>>(e2w2, 16, 3,  0,  chunk(12));
  pack_B_kernel<<<1, 32, 0, stream>>>(n2w0, 19, 16, 0,  chunk(13));
  pack_B_kernel<<<1, 32, 0, stream>>>(n2w1, 16, 16, 0,  chunk(14));
  pack_B_kernel<<<1, 32, 0, stream>>>(n2w2, 16, 3,  0,  chunk(15));

  const int eTiles = N_EDGES / 16;           // 200000 (exact)
  const int nTiles = N_NODES / 16;           // 6250   (exact)
  const int eBlocks = (eTiles + 7) / 8;      // 25000
  const int nBlocks = (nTiles + 7) / 8;      // 782
  const size_t aggBytes = (size_t)N_NODES * 32 * sizeof(float);

  // ---- conv 1 ----
  hipMemsetAsync(agg, 0, aggBytes, stream);
  edge_conv_kernel<12, 6, 32, 2, true, 16><<<eBlocks, 256, 0, stream>>>(
      x_node, x_edge, src, dst,
      (const v16bf*)chunk(0), e0b0, (const v16bf*)chunk(1), e0b1, nullptr, nullptr,
      agg, xe0, eTiles);
  node_conv_kernel<28, 2, true, 16><<<nBlocks, 256, 0, stream>>>(
      agg, (const v16bf*)chunk(2), n0b0, (const v16bf*)chunk(3), n0b1,
      nullptr, nullptr, xn0, 16, nTiles);

  // ---- conv 2 ----
  hipMemsetAsync(agg, 0, aggBytes, stream);
  edge_conv_kernel<16, 16, 64, 2, true, 16><<<eBlocks, 256, 0, stream>>>(
      xn0, xe0, src, dst,
      (const v16bf*)chunk(4), e1b0, (const v16bf*)chunk(6), e1b1, nullptr, nullptr,
      agg, xe1, eTiles);
  node_conv_kernel<32, 2, true, 16><<<nBlocks, 256, 0, stream>>>(
      agg, (const v16bf*)chunk(7), n1b0, (const v16bf*)chunk(8), n1b1,
      nullptr, nullptr, xn1, 16, nTiles);

  // ---- conv 3 (final linear, no ReLU) ----
  hipMemsetAsync(agg, 0, aggBytes, stream);
  edge_conv_kernel<16, 16, 64, 3, false, 3><<<eBlocks, 256, 0, stream>>>(
      xn1, xe1, src, dst,
      (const v16bf*)chunk(9), e2b0, (const v16bf*)chunk(11), e2b1,
      (const v16bf*)chunk(12), e2b2, agg, out_edge, eTiles);
  node_conv_kernel<19, 3, false, 3><<<nBlocks, 256, 0, stream>>>(
      agg, (const v16bf*)chunk(13), n2b0, (const v16bf*)chunk(14), n2b1,
      (const v16bf*)chunk(15), n2b2, out_node, 3, nTiles);
}
